// SimpleLstmEncoder_25340307046419
// MI455X (gfx1250) — matrix operand order repaired and checked by reading, not compiled, not run on hardware
//
#include <hip/hip_runtime.h>

// ---------------- sizes ----------------
#define B_   64
#define T_   512
#define F_   64
#define E_   256
#define H_   512
#define G_   2048   // 4*H
#define O_   128
#define NWG  16     // workgroups in recurrent persistent kernel
#define COLS 32     // h-columns owned per WG  (H_/NWG)
#define GN   128    // gate columns per WG     (4*COLS)

typedef __attribute__((ext_vector_type(16))) __bf16 v16bf;
typedef __attribute__((ext_vector_type(8)))  __bf16 v8bf;
typedef __attribute__((ext_vector_type(8)))  float  v8f;
typedef __attribute__((ext_vector_type(4)))  unsigned v4u;
typedef __attribute__((ext_vector_type(8)))  int      v8i;
typedef __attribute__((ext_vector_type(4)))  int      v4i;

// ---------------- WMMA fragment helpers (wave32, gfx1250 layouts) ----------------
// A tile 16(M)x32(K) bf16, row-major source with stride ld.
// Lane L: M=L%16, half=L/16. VGPR v holds K pair: half*16 + 2v.
__device__ __forceinline__ v16bf load_a(const __bf16* src, int ld) {
  int lane = threadIdx.x & 31;
  int m    = lane & 15;
  int half = lane >> 4;
  const __bf16* row = src + m * ld;
  v8bf lo = *(const v8bf*)(row + half * 16);       // K = half*16 .. +7
  v8bf hi = *(const v8bf*)(row + half * 16 + 8);   // K = half*16+8 .. +15
  v16bf a;
#pragma unroll
  for (int i = 0; i < 8; ++i) { a[i] = lo[i]; a[8 + i] = hi[i]; }
  return a;
}

// B tile 32(K)x16(N) where B[k][n] = w[n*ld + k]  (i.e. B = W^T, W row-major).
// Lane L: N=L%16, half=L/16. VGPR v holds K pair: half*16 + 2v.
__device__ __forceinline__ v16bf load_bt(const __bf16* w, int ld) {
  int lane = threadIdx.x & 31;
  int n    = lane & 15;
  int half = lane >> 4;
  const __bf16* row = w + n * ld + half * 16;      // 32B contiguous
  v16bf b;
#pragma unroll
  for (int i = 0; i < 16; ++i) b[i] = row[i];
  return b;
}

// ---------------- Tensor Data Mover (TDM) helpers ----------------
// Low 32 bits of a generic pointer to LDS == LDS byte address (ISA 10.2 aperture rule).
__device__ __forceinline__ unsigned lds_off(const void* p) {
  return (unsigned)(uintptr_t)p;
}

// TDM load of a bf16 tile, tile==tensor, up to 3D:
//   addr = base + 2*(x + y*s0 + z*s1),  x<d0, y<d1, z<d2 (d2==0 -> 2D/1D)
// Packs contiguously into LDS at ldsAddr. Dims/strides in elements.
__device__ __forceinline__ void tdm_load_bf16(unsigned ldsAddr, const void* gptr,
                                              unsigned d0, unsigned d1, unsigned d2,
                                              unsigned long long s0,
                                              unsigned long long s1) {
  unsigned long long ga = (unsigned long long)(uintptr_t)gptr;
  v4u g0; v8i g1; v4i g2, g3;
  // --- group 0: count=1, lds_addr, global_addr, type=2 ("image")
  g0[0] = 1u;
  g0[1] = ldsAddr;
  g0[2] = (unsigned)(ga & 0xffffffffu);
  g0[3] = (unsigned)((ga >> 32) & 0x1ffffffu) | 0x80000000u;
  // --- group 1: wg_mask=0, data_size=1 (2 bytes), dims/tiles/strides
  g1[0] = (int)(1u << 16);
  g1[1] = (int)(d0 << 16);                                   // tensor_dim0[15:0] @bit48
  g1[2] = (int)((d0 >> 16) | (d1 << 16));                    // dim0 hi | dim1 lo
  g1[3] = (int)((d1 >> 16) | (d0 << 16));                    // dim1 hi | tile_dim0
  g1[4] = (int)((d1 & 0xffffu) | (d2 << 16));                // tile_dim1 | tile_dim2
  g1[5] = (int)(unsigned)(s0 & 0xffffffffu);                 // dim0_stride lo32
  g1[6] = (int)((unsigned)((s0 >> 32) & 0xffffu) |
                ((unsigned)(s1 & 0xffffu) << 16));           // dim0_stride hi | dim1_stride lo
  g1[7] = (int)(unsigned)((s1 >> 16) & 0xffffffffu);         // dim1_stride [47:16]
  // --- group 2: tensor_dim2; group 3: unused
  g2[0] = (int)d2; g2[1] = 0; g2[2] = 0; g2[3] = 0;
  g3[0] = 0; g3[1] = 0; g3[2] = 0; g3[3] = 0;
#if defined(__clang_major__) && __clang_major__ >= 23
  v8i g4 = {0, 0, 0, 0, 0, 0, 0, 0};
  __builtin_amdgcn_tensor_load_to_lds(g0, g1, g2, g3, g4, 0);
#else
  __builtin_amdgcn_tensor_load_to_lds(g0, g1, g2, g3, 0);
#endif
}

// ---------------- grid barrier (persistent-kernel sync) ----------------
__device__ __forceinline__ void grid_barrier(unsigned* bar, unsigned* epoch) {
  __threadfence();            // release h writes device-wide
  __syncthreads();
  ++*epoch;
  if (threadIdx.x == 0) {
    unsigned* cnt = bar;       // separate cachelines
    unsigned* gen = bar + 64;
    unsigned old = __hip_atomic_fetch_add(cnt, 1u, __ATOMIC_ACQ_REL, __HIP_MEMORY_SCOPE_AGENT);
    if (old == NWG - 1) {
      __hip_atomic_store(cnt, 0u, __ATOMIC_RELAXED, __HIP_MEMORY_SCOPE_AGENT);
      __hip_atomic_fetch_add(gen, 1u, __ATOMIC_RELEASE, __HIP_MEMORY_SCOPE_AGENT);
    }
    while (__hip_atomic_load(gen, __ATOMIC_ACQUIRE, __HIP_MEMORY_SCOPE_AGENT) < *epoch) {
      __builtin_amdgcn_s_sleep(2);
    }
  }
  __syncthreads();
}

// ---------------- prep kernels ----------------
__global__ void k_cvt(const float* __restrict__ in, __bf16* __restrict__ out, int n) {
  int i = blockIdx.x * blockDim.x + threadIdx.x;
  if (i < n) out[i] = (__bf16)in[i];
}

// W_comb[r,f] = sum_e W_ih[r,e] * W_emb[e,f]   (2048x64, bf16 out)
__global__ void k_comb(const float* __restrict__ wih, const float* __restrict__ wemb,
                       __bf16* __restrict__ wc) {
  int t = blockIdx.x * blockDim.x + threadIdx.x;
  if (t >= G_ * F_) return;
  int r = t / F_, f = t % F_;
  float s = 0.f;
  for (int e = 0; e < E_; ++e) s += wih[(size_t)r * E_ + e] * wemb[(size_t)e * F_ + f];
  wc[t] = (__bf16)s;
}

// bias_eff[r] = b_ih[r] + b_hh[r] + sum_e W_ih[r,e]*b_emb[e]; also zero barrier state.
__global__ void k_bias(const float* __restrict__ wih, const float* __restrict__ bemb,
                       const float* __restrict__ bih, const float* __restrict__ bhh,
                       float* __restrict__ bias, unsigned* __restrict__ bar) {
  int t = blockIdx.x * blockDim.x + threadIdx.x;
  if (t < 128) bar[t] = 0u;          // cnt + gen re-zeroed every launch (deterministic)
  if (t >= G_) return;
  float s = bih[t] + bhh[t];
  for (int e = 0; e < E_; ++e) s += wih[(size_t)t * E_ + e] * bemb[e];
  bias[t] = s;
}

// ---------------- fused input GEMM: xg[32768,2048] = x_bf16 @ W_comb^T + bias ----------------
__global__ __launch_bounds__(256) void k_xg(const __bf16* __restrict__ xbf,
                                            const __bf16* __restrict__ wc,
                                            const float* __restrict__ bias,
                                            float* __restrict__ xg) {
  int gw = blockIdx.x * 8 + (threadIdx.x >> 5); // global wave id
  int mt = gw >> 5;                              // 0..2047 (16-row tiles of 32768)
  int ng = gw & 31;                              // 0..31   (64-col groups of 2048)
  int m0 = mt * 16, n0 = ng * 64;
  v8f acc[4] = {};
#pragma unroll
  for (int kt = 0; kt < 2; ++kt) {
    v16bf a = load_a(xbf + (size_t)m0 * F_ + kt * 32, F_);
#pragma unroll
    for (int s = 0; s < 4; ++s) {
      v16bf b = load_bt(wc + (size_t)(n0 + 16 * s) * F_ + kt * 32, F_);
      acc[s] = __builtin_amdgcn_wmma_f32_16x16x32_bf16(false, a, false, b,
                                                       (short)0, acc[s], false, false);
    }
  }
  int lane = threadIdx.x & 31;
  int n = lane & 15, mb = (lane >> 4) * 8;
#pragma unroll
  for (int s = 0; s < 4; ++s) {
    int col0 = n0 + 16 * s;
#pragma unroll
    for (int r = 0; r < 8; ++r)   // xg is stream-once: keep it out of L2 (NT store)
      __builtin_nontemporal_store(acc[s][r] + bias[col0 + n],
                                  &xg[(size_t)(m0 + mb + r) * G_ + col0 + n]);
  }
}

// ---------------- persistent recurrent LSTM kernel ----------------
// 16 WGs, each owns 32 h-columns (=> 128 gate rows of W_hh, LDS-resident all scan).
// TDM stages W_hh slice (once) and h_t (per step) into LDS; compute overlaps via waves.
__global__ __launch_bounds__(256) void k_lstm(const float* __restrict__ xg,
                                              const __bf16* __restrict__ whh,
                                              __bf16* __restrict__ hbuf,   // 2 x [64,512]
                                              float* __restrict__ pooled,
                                              unsigned* __restrict__ bar) {
  __shared__ __bf16 sW[GN * H_];     // 128 KB: this WG's W_hh rows (bf16)
  __shared__ __bf16 sH[B_ * H_];     //  64 KB: staged h_t (full, bf16)
  __shared__ float  sG[B_ * GN];     //  32 KB: gate pre-activations
  __shared__ float  sC[B_ * COLS];   //   8 KB: cell state (persistent)
  __shared__ float  sS[B_ * COLS];   //   8 KB: running sum of h (pooling)
  const int tid  = threadIdx.x;
  const int wg   = blockIdx.x;
  const int wid  = tid >> 5;
  const int lane = tid & 31;
  unsigned epoch = 0;

  // One TDM 3D load pulls the whole 128 KB W_hh slice:
  // x = 512-elem row, y = 32 rows (contiguous, stride H_), z = 4 gate blocks (stride H_*H_).
  if (wid == 0) {
    tdm_load_bf16(lds_off(sW), whh + (size_t)wg * COLS * H_,
                  H_, COLS, 4,
                  (unsigned long long)H_, (unsigned long long)H_ * H_);
  }
  for (int i = tid; i < B_ * COLS; i += 256) { sC[i] = 0.f; sS[i] = 0.f; }
  for (int i = tid; i < B_ * COLS; i += 256) {         // zero own cols of h buffer 0
    int b = i >> 5, j = i & 31;
    hbuf[(size_t)b * H_ + wg * COLS + j] = (__bf16)0.f;
  }
  if (wid == 0) __builtin_amdgcn_s_wait_tensorcnt(0);
  grid_barrier(bar, &epoch);   // all h0 columns + weights visible

  for (int s = 0; s < T_; ++s) {
    const __bf16* hsrc = hbuf + (size_t)(s & 1) * B_ * H_;
    __bf16*       hdst = hbuf + (size_t)((s + 1) & 1) * B_ * H_;

    // Stage h_t into LDS via TDM: one 1D tile of 32768 bf16 (64 KB), DMA'd by wave 0.
    if (wid == 0) {
      tdm_load_bf16(lds_off(sH), hsrc, B_ * H_, 1, 0,
                    (unsigned long long)(B_ * H_), 0ull);
      __builtin_amdgcn_s_wait_tensorcnt(0);
    }
    __syncthreads();

    // GEMM: gates64x128 = h(64x512,bf16) @ sW^T; wave wid owns 16-col tile, 4 row tiles
    const int nbase = wid * 16;
    v8f acc[4] = {};
#pragma unroll 2
    for (int kt = 0; kt < 16; ++kt) {
      v16bf bf = load_bt(sW + (size_t)nbase * H_ + kt * 32, H_);
#pragma unroll
      for (int mt = 0; mt < 4; ++mt) {
        v16bf af = load_a(sH + (size_t)mt * 16 * H_ + kt * 32, H_);
        acc[mt] = __builtin_amdgcn_wmma_f32_16x16x32_bf16(false, af, false, bf,
                                                          (short)0, acc[mt], false, false);
      }
    }
    int n = lane & 15, mb = (lane >> 4) * 8;
#pragma unroll
    for (int mt = 0; mt < 4; ++mt)
#pragma unroll
      for (int r = 0; r < 8; ++r)
        sG[(size_t)(mt * 16 + mb + r) * GN + nbase + n] = acc[mt][r];
    __syncthreads();

    // elementwise gate math + state update (PyTorch gate order i,f,g,o)
    for (int p = tid; p < B_ * COLS; p += 256) {
      int b = p >> 5, j = p & 31;
      size_t xr = ((size_t)b * T_ + s) * G_ + wg * COLS + j;
      float gi = sG[(size_t)b * GN + 0 * COLS + j] + __builtin_nontemporal_load(&xg[xr + 0 * H_]);
      float gf = sG[(size_t)b * GN + 1 * COLS + j] + __builtin_nontemporal_load(&xg[xr + 1 * H_]);
      float gg = sG[(size_t)b * GN + 2 * COLS + j] + __builtin_nontemporal_load(&xg[xr + 2 * H_]);
      float go = sG[(size_t)b * GN + 3 * COLS + j] + __builtin_nontemporal_load(&xg[xr + 3 * H_]);
      // prefetch next timestep's xg slice while the scan is barrier-bound
      __builtin_prefetch(&xg[xr + (size_t)G_], 0, 1);
      gi = 1.f / (1.f + __expf(-gi));
      gf = 1.f / (1.f + __expf(-gf));
      gg = tanhf(gg);
      go = 1.f / (1.f + __expf(-go));
      float c = gf * sC[p] + gi * gg;
      sC[p] = c;
      float h = go * tanhf(c);
      sS[p] += h;
      hdst[(size_t)b * H_ + wg * COLS + j] = (__bf16)h;
    }
    grid_barrier(bar, &epoch);   // h_{t+1} complete everywhere
  }

  for (int p = tid; p < B_ * COLS; p += 256) {
    int b = p >> 5, j = p & 31;
    pooled[(size_t)b * H_ + wg * COLS + j] = sS[p] * (1.0f / T_);
  }
}

// ---------------- FC head: out[64,128] = pooled @ W_fc^T + b_fc ----------------
__global__ void k_fc(const float* __restrict__ pooled, const float* __restrict__ wfc,
                     const float* __restrict__ bfc, float* __restrict__ out) {
  int t = blockIdx.x * blockDim.x + threadIdx.x;
  if (t >= B_ * O_) return;
  int b = t / O_, o = t % O_;
  float s = bfc[o];
  const float* pr = pooled + (size_t)b * H_;
  const float* wr = wfc + (size_t)o * H_;
  for (int h = 0; h < H_; ++h) s += pr[h] * wr[h];
  out[t] = s;
}

// ---------------- launcher ----------------
extern "C" void kernel_launch(void* const* d_in, const int* in_sizes, int n_in,
                              void* d_out, int out_size, void* d_ws, size_t ws_size,
                              hipStream_t stream) {
  const float* x    = (const float*)d_in[0];
  const float* Wemb = (const float*)d_in[1];
  const float* bemb = (const float*)d_in[2];
  const float* Wih  = (const float*)d_in[3];
  const float* Whh  = (const float*)d_in[4];
  const float* bih  = (const float*)d_in[5];
  const float* bhh  = (const float*)d_in[6];
  const float* Wfc  = (const float*)d_in[7];
  const float* bfc  = (const float*)d_in[8];
  float* out = (float*)d_out;

  char* ws = (char*)d_ws;
  size_t off = 0;
  float*  xg     = (float*)(ws + off);  off += (size_t)B_ * T_ * G_ * 4;   // 268 MB
  __bf16* xbf    = (__bf16*)(ws + off); off += (size_t)B_ * T_ * F_ * 2;   // 4.2 MB
  __bf16* wc     = (__bf16*)(ws + off); off += (size_t)G_ * F_ * 2;        // 256 KB
  __bf16* whhb   = (__bf16*)(ws + off); off += (size_t)G_ * H_ * 2;        // 2 MB
  float*  bias   = (float*)(ws + off);  off += (size_t)G_ * 4;             // 8 KB
  __bf16* hbuf   = (__bf16*)(ws + off); off += (size_t)2 * B_ * H_ * 2;    // 128 KB (ping-pong)
  float*  pooled = (float*)(ws + off);  off += (size_t)B_ * H_ * 4;        // 128 KB
  unsigned* bar  = (unsigned*)(ws + off); off += 512;

  k_cvt <<<(B_ * T_ * F_ + 255) / 256, 256, 0, stream>>>(x, xbf, B_ * T_ * F_);
  k_cvt <<<(G_ * H_ + 255) / 256, 256, 0, stream>>>(Whh, whhb, G_ * H_);
  k_comb<<<(G_ * F_ + 255) / 256, 256, 0, stream>>>(Wih, Wemb, wc);
  k_bias<<<(G_ + 255) / 256, 256, 0, stream>>>(Wih, bemb, bih, bhh, bias, bar);
  k_xg  <<<(B_ * T_ / 16) * (G_ / 64) / 8, 256, 0, stream>>>(xbf, wc, bias, xg);
  k_lstm<<<NWG, 256, 0, stream>>>(xg, whhb, hbuf, pooled, bar);
  k_fc  <<<(B_ * O_ + 255) / 256, 256, 0, stream>>>(pooled, Wfc, bfc, out);
}
